// GraphTransformer_22213570854985
// MI455X (gfx1250) — matrix operand order repaired and checked by reading, not compiled, not run on hardware
//
#include <hip/hip_runtime.h>
#include <hip/hip_bf16.h>

// ---------------------------------------------------------------------------
// Types for CDNA5 WMMA
// ---------------------------------------------------------------------------
typedef __attribute__((ext_vector_type(16))) __bf16 v16bf;
typedef __attribute__((ext_vector_type(8)))  float  v8f;

union FragU {
    uint4 u[2];
    v16bf v;
};

__device__ __forceinline__ unsigned short f32_to_bf16_rne(float f) {
    unsigned u = __float_as_uint(f);
    unsigned r = u + 0x7FFFu + ((u >> 16) & 1u);
    return (unsigned short)(r >> 16);
}

// order-preserving f32 -> u32 map (for atomicMax-based segment max)
__device__ __forceinline__ unsigned f32_ord(float f) {
    unsigned u = __float_as_uint(f);
    return (u & 0x80000000u) ? ~u : (u | 0x80000000u);
}
__device__ __forceinline__ float ord_f32(unsigned u) {
    unsigned v = (u & 0x80000000u) ? (u & 0x7FFFFFFFu) : ~u;
    return __uint_as_float(v);
}

// gfx1250 async DMA: copy 16 bytes global -> LDS, tracked by ASYNCcnt.
// (inline asm: portable across ROCm7.2 / amdgpu-toolchain builtin arities)
__device__ __forceinline__ void async_copy_b128(unsigned lds_byte_addr,
                                                const void* gaddr) {
    asm volatile("global_load_async_to_lds_b128 %0, %1, off"
                 :: "v"(lds_byte_addr), "v"((unsigned long long)(size_t)gaddr)
                 : "memory");
}
__device__ __forceinline__ void wait_async0() {
    asm volatile("s_wait_asynccnt 0x0" ::: "memory");
}

// ---------------------------------------------------------------------------
// fp32 -> bf16 conversions (plain and transposing, for weights)
// ---------------------------------------------------------------------------
__global__ void cvt_bf16_kernel(const float* __restrict__ in,
                                unsigned short* __restrict__ out, int n) {
    int t = blockIdx.x * blockDim.x + threadIdx.x;
    if (t < n) out[t] = f32_to_bf16_rne(in[t]);
}

// in: [K][N] fp32 row-major  ->  out: [N][K] bf16 (transposed weights)
__global__ void cvt_bf16_t_kernel(const float* __restrict__ in,
                                  unsigned short* __restrict__ out,
                                  int K, int N) {
    int t = blockIdx.x * blockDim.x + threadIdx.x;
    if (t >= K * N) return;
    int k = t / N, n = t - k * N;
    out[(size_t)n * K + k] = f32_to_bf16_rne(in[t]);
}

// ---------------------------------------------------------------------------
// WMMA GEMM:  C[M,Nout] = A[M,K](bf16) * B[K,Nout](bf16, given as Bt=[N][K]) + bias
//
// 256 threads (8 waves). Block tile: 128 rows x NTILES*16 cols.
// Double-buffered LDS tiles filled by global_load_async_to_lds_b128:
//   A tile [128][32] bf16 (8 KB), B tile [Npad][32] bf16 (<= 8 KB), x2 buffers.
// Pipeline per K-step: issue async loads for tile k+1 -> alt buffers,
// run ds_load_b128 fragment loads + v_wmma on tile k, s_wait_asynccnt 0,
// one workgroup barrier.  Out-of-range rows/cols skip their copies (their
// LDS garbage only reaches output rows/cols that are never stored).
// All LDS addressing uses integer offsets off one extern __shared__ base
// (arrays of LDS-derived pointers break ld.lld static initializers).
// ---------------------------------------------------------------------------
#define KSTEP 32

template <int NTILES>
__global__ __launch_bounds__(256)
void gemm_bias_wmma(const unsigned short* __restrict__ Abf,  // [M][K] bf16
                    const unsigned short* __restrict__ Bt,   // [Nout][K] bf16
                    const float* __restrict__ bias,
                    float* __restrict__ C,
                    int M, int K, int Nout) {
    constexpr int NPAD   = NTILES * 16;
    constexpr int ATILE  = 128 * KSTEP;   // elements per A buffer
    constexpr int BTILE  = NPAD * KSTEP;  // elements per B buffer
    extern __shared__ unsigned short lds_u16[];

    const int tid   = threadIdx.x;
    const int wave  = tid >> 5;
    const int lane  = tid & 31;
    const int lhalf = lane >> 4;
    const int lmod  = lane & 15;
    const int row_base = blockIdx.x * 128;
    const int ksteps = K / KSTEP;

    v8f acc[NTILES];
#pragma unroll
    for (int t = 0; t < NTILES; ++t) acc[t] = (v8f){0, 0, 0, 0, 0, 0, 0, 0};

    auto stage = [&](int kt, int buf) {
        const int kk = kt * KSTEP;
        const int abase = buf * ATILE;
        const int bbase = 2 * ATILE + buf * BTILE;
        // A tile: 128 rows x 4 x 16B chunks
        for (int c = tid; c < 128 * 4; c += 256) {
            int r = c >> 2, ch = c & 3;
            int row = row_base + r;
            if (row < M) {
                unsigned dl =
                    (unsigned)(size_t)(lds_u16 + abase + r * KSTEP + ch * 8);
                async_copy_b128(dl, Abf + (size_t)row * K + kk + ch * 8);
            }
        }
        // B tile: Npad cols x 4 x 16B chunks (Bt is [N][K] so K is contiguous)
        for (int c = tid; c < NPAD * 4; c += 256) {
            int n = c >> 2, ch = c & 3;
            if (n < Nout) {
                unsigned dl =
                    (unsigned)(size_t)(lds_u16 + bbase + n * KSTEP + ch * 8);
                async_copy_b128(dl, Bt + (size_t)n * K + kk + ch * 8);
            }
        }
    };

    stage(0, 0);
    wait_async0();
    __syncthreads();

    for (int kt = 0; kt < ksteps; ++kt) {
        const int buf = kt & 1;
        if (kt + 1 < ksteps) stage(kt + 1, buf ^ 1);

        const int abase = buf * ATILE;
        const int bbase = 2 * ATILE + buf * BTILE;

        // A fragment (16-bit A 16x32 layout): lane row = wave*16+lmod;
        // V0..3 -> K = 8*lhalf+0..7 ; V4..7 -> K = 16+8*lhalf+0..7
        FragU af;
        {
            const unsigned short* ap =
                lds_u16 + abase + (wave * 16 + lmod) * KSTEP;
            af.u[0] = *(const uint4*)(ap + 8 * lhalf);
            af.u[1] = *(const uint4*)(ap + 16 + 8 * lhalf);
        }
#pragma unroll
        for (int nt = 0; nt < NTILES; ++nt) {
            // B fragment (32x16): lane col = nt*16+lmod, K = 8*lhalf+0..7, 16+8*lhalf+0..7
            FragU bf;
            const unsigned short* bp =
                lds_u16 + bbase + (nt * 16 + lmod) * KSTEP + 8 * lhalf;
            bf.u[0] = *(const uint4*)(bp);
            bf.u[1] = *(const uint4*)(bp + 16);
            acc[nt] = __builtin_amdgcn_wmma_f32_16x16x32_bf16(
                false, af.v, false, bf.v, (short)0, acc[nt], false, false);
        }

        wait_async0();
        __syncthreads();
    }

    // C/D layout: lane lmod = col-in-tile, VGPR r -> row = 8*lhalf + r
#pragma unroll
    for (int nt = 0; nt < NTILES; ++nt) {
        int col = nt * 16 + lmod;
#pragma unroll
        for (int r = 0; r < 8; ++r) {
            int row = row_base + wave * 16 + 8 * lhalf + r;
            if (row < M && col < Nout)
                C[(size_t)row * Nout + col] = acc[nt][r] + bias[col];
        }
    }
}

// ---------------------------------------------------------------------------
// Edge phase kernels (segment softmax + aggregation)
// ---------------------------------------------------------------------------
__global__ void init_ms_kernel(unsigned* __restrict__ m_u,
                               float* __restrict__ s, int n) {
    int t = blockIdx.x * blockDim.x + threadIdx.x;
    if (t < n) { m_u[t] = 0x007FFFFFu /* ord(-inf) */; s[t] = 0.0f; }
}

// pass 1: alpha[e,h] = (q[dst,h,:] . k[src,h,:]) / sqrt(dh); segment max
__global__ void edge_alpha_kernel(const float* __restrict__ q,
                                  const float* __restrict__ k,
                                  const int* __restrict__ src,
                                  const int* __restrict__ dst,
                                  float* __restrict__ alpha,
                                  unsigned* __restrict__ m_u,
                                  int E, int D) {
    int t = blockIdx.x * blockDim.x + threadIdx.x;
    if (t >= E * 8) return;
    int e = t >> 3, hh = t & 7;
    int sn = src[e], dn = dst[e];
    int dh = D >> 3;
    const float* qp = q + (size_t)dn * D + hh * dh;
    const float* kp = k + (size_t)sn * D + hh * dh;
    float acc = 0.0f;
    for (int j = 0; j < dh; ++j) acc += qp[j] * kp[j];
    acc *= rsqrtf((float)dh);
    alpha[t] = acc;
    atomicMax(&m_u[dn * 8 + hh], f32_ord(acc));
}

// pass 2: alpha := exp(alpha - m[dst]); segment sum
__global__ void edge_exp_kernel(const int* __restrict__ dst,
                                float* __restrict__ alpha,
                                const unsigned* __restrict__ m_u,
                                float* __restrict__ s, int E) {
    int t = blockIdx.x * blockDim.x + threadIdx.x;
    if (t >= E * 8) return;
    int e = t >> 3, hh = t & 7;
    int dn = dst[e];
    float mf = ord_f32(m_u[dn * 8 + hh]);
    float ev = __expf(alpha[t] - mf);
    alpha[t] = ev;
    atomicAdd(&s[dn * 8 + hh], ev);
}

// pass 3: out[dst, i] += (alpha / (s[dst]+eps)) * v[src, i]
__global__ void edge_agg_kernel(const int* __restrict__ src,
                                const int* __restrict__ dst,
                                const float* __restrict__ alpha,
                                const float* __restrict__ s,
                                const float* __restrict__ v,
                                float* __restrict__ out,
                                int E, int log2D) {
    int t = blockIdx.x * blockDim.x + threadIdx.x;
    int D = 1 << log2D;
    if (t >= (E << log2D)) return;
    int e = t >> log2D;
    int i = t & (D - 1);
    int hh = i >> (log2D - 3);   // dh = D/8
    int sn = src[e], dn = dst[e];
    float a = alpha[e * 8 + hh] / (s[dn * 8 + hh] + 1e-16f);
    atomicAdd(&out[(size_t)dn * D + i], a * v[(size_t)sn * D + i]);
}

// ---------------------------------------------------------------------------
// ReLU + BatchNorm (eval).  bn = bn(relu(in)).
// Always emits bf16 bn-output (next layer's GEMM A operand);
// optionally emits f32 relu(in) and f32 bn (for the h4 / hbn4 outputs).
// ---------------------------------------------------------------------------
__global__ void relu_bn_kernel(const float* __restrict__ in,
                               unsigned short* __restrict__ out_bf,
                               float* __restrict__ out_relu,
                               float* __restrict__ out_bn,
                               const float* __restrict__ g,
                               const float* __restrict__ b,
                               const float* __restrict__ m,
                               const float* __restrict__ vv,
                               int n, int D) {
    int t = blockIdx.x * blockDim.x + threadIdx.x;
    if (t >= n) return;
    int c = t & (D - 1);
    float x = in[t];
    x = x > 0.0f ? x : 0.0f;
    float bn = (x - m[c]) * rsqrtf(vv[c] + 1e-5f) * g[c] + b[c];
    out_bf[t] = f32_to_bf16_rne(bn);
    if (out_relu) out_relu[t] = x;
    if (out_bn)   out_bn[t]   = bn;
}

// ---------------------------------------------------------------------------
// Host orchestration
// ---------------------------------------------------------------------------
extern "C" void kernel_launch(void* const* d_in, const int* in_sizes, int n_in,
                              void* d_out, int out_size, void* d_ws, size_t ws_size,
                              hipStream_t stream) {
    (void)n_in; (void)out_size; (void)ws_size;
    const int GENE = 256, HID = 128, HID2 = 64;
    const int Nn = in_sizes[0] / GENE;     // 100000
    const int E  = in_sizes[1] / 2;        // 800000

    const float* x  = (const float*)d_in[0];
    const int*   ei = (const int*)d_in[1];
    const int* src = ei;
    const int* dst = ei + E;

    // input index map (flattened setup_inputs dict order)
    // conv c (0..3): Wq=2+8c, bq=3+8c, Wk=4+8c, bk=5+8c, Wv=6+8c, bv=7+8c,
    //                Ws=8+8c, bs=9+8c
    // bn  c (0..3): g=34+4c, b=35+4c, m=36+4c, v=37+4c
    // Wc=50, bc=51

    // workspace layout
    const size_t NH = (size_t)Nn * HID;
    float* q   = (float*)d_ws;
    float* k   = q + NH;
    float* v   = k + NH;
    float* out = v + NH;
    float* alpha = out + NH;               // E*8
    unsigned* m_u = (unsigned*)(alpha + (size_t)E * 8);
    float* s = (float*)(m_u + (size_t)Nn * 8);
    unsigned short* abf = (unsigned short*)(s + (size_t)Nn * 8); // [Nn][<=256] bf16
    unsigned short* wbf = abf + (size_t)Nn * GENE;               // transposed weights

    // bf16 transposed weight slots: conv1 Wq,Wk,Wv,Ws | conv2 | conv3 | conv4 | Wc
    const int wK[17]  = {256,256,256,256, 128,128,128,128, 128,128,128,128,
                         128,128,128,128, 64};
    const int wN[17]  = {128,128,128,128, 128,128,128,128, 128,128,128,128,
                         64,64,64,64, 10};
    const int widx[17] = {2,4,6,8, 10,12,14,16, 18,20,22,24, 26,28,30,32, 50};
    size_t woff[17];
    {
        size_t o = 0;
        for (int i = 0; i < 17; ++i) { woff[i] = o; o += (size_t)wK[i] * wN[i]; }
        for (int i = 0; i < 17; ++i) {
            int n = wK[i] * wN[i];
            cvt_bf16_t_kernel<<<(n + 255) / 256, 256, 0, stream>>>(
                (const float*)d_in[widx[i]], wbf + woff[i], wK[i], wN[i]);
        }
    }

    auto gemm = [&](const unsigned short* A, const unsigned short* Bt,
                    const float* bias, float* C, int M, int K, int Nout) {
        int blocks = (M + 127) / 128;
        if (Nout > 64) {
            size_t sh = (size_t)(2 * 128 * KSTEP + 2 * 128 * KSTEP) * 2;
            gemm_bias_wmma<8><<<blocks, 256, sh, stream>>>(A, Bt, bias, C, M, K, Nout);
        } else if (Nout > 16) {
            size_t sh = (size_t)(2 * 128 * KSTEP + 2 * 64 * KSTEP) * 2;
            gemm_bias_wmma<4><<<blocks, 256, sh, stream>>>(A, Bt, bias, C, M, K, Nout);
        } else {
            size_t sh = (size_t)(2 * 128 * KSTEP + 2 * 16 * KSTEP) * 2;
            gemm_bias_wmma<1><<<blocks, 256, sh, stream>>>(A, Bt, bias, C, M, K, Nout);
        }
    };

    // layer-1 input: x -> bf16 activations
    {
        int n = Nn * GENE;
        cvt_bf16_kernel<<<(n + 255) / 256, 256, 0, stream>>>(x, abf, n);
    }

    // d_out slices: logits [Nn*10] | h4 [Nn*64] | hbn4 [Nn*64]
    float* logits = (float*)d_out;
    float* h4     = logits + (size_t)Nn * 10;
    float* hbn4   = h4 + (size_t)Nn * HID2;

    const int EH  = E * 8;
    const int NHb = Nn * 8;

    for (int c = 0; c < 4; ++c) {
        int K  = (c == 0) ? GENE : HID;
        int D  = (c == 3) ? HID2 : HID;
        int wb = 4 * c;
        int bi = 2 + 8 * c;   // Wq idx; biases at +1,+3,+5,+7

        gemm(abf, wbf + woff[wb + 0], (const float*)d_in[bi + 1], q,   Nn, K, D);
        gemm(abf, wbf + woff[wb + 1], (const float*)d_in[bi + 3], k,   Nn, K, D);
        gemm(abf, wbf + woff[wb + 2], (const float*)d_in[bi + 5], v,   Nn, K, D);
        gemm(abf, wbf + woff[wb + 3], (const float*)d_in[bi + 7], out, Nn, K, D);

        init_ms_kernel<<<(NHb + 255) / 256, 256, 0, stream>>>(m_u, s, NHb);
        edge_alpha_kernel<<<(EH + 255) / 256, 256, 0, stream>>>(
            q, k, src, dst, alpha, m_u, E, D);
        edge_exp_kernel<<<(EH + 255) / 256, 256, 0, stream>>>(
            dst, alpha, m_u, s, E);
        int log2D = (D == 128) ? 7 : 6;
        int total = E << log2D;
        edge_agg_kernel<<<(total + 255) / 256, 256, 0, stream>>>(
            src, dst, alpha, s, v, out, E, log2D);

        const float* bg = (const float*)d_in[34 + 4 * c];
        const float* bb = (const float*)d_in[35 + 4 * c];
        const float* bm = (const float*)d_in[36 + 4 * c];
        const float* bv = (const float*)d_in[37 + 4 * c];
        int nelem = Nn * D;
        if (c < 3) {
            relu_bn_kernel<<<(nelem + 255) / 256, 256, 0, stream>>>(
                out, abf, nullptr, nullptr, bg, bb, bm, bv, nelem, D);
        } else {
            relu_bn_kernel<<<(nelem + 255) / 256, 256, 0, stream>>>(
                out, abf, h4, hbn4, bg, bb, bm, bv, nelem, D);
        }
    }

    // classifier: logits = hbn4 @ Wc + bc   (abf holds bf16(hbn4))
    gemm(abf, wbf + woff[16], (const float*)d_in[51], logits, Nn, HID2, 10);
}
